// AUAttnProcessor_34952443855355
// MI455X (gfx1250) — matrix operand order repaired and checked by reading, not compiled, not run on hardware
//
#include <hip/hip_runtime.h>
#include <hip/hip_bf16.h>

typedef __attribute__((ext_vector_type(16))) _Float16 v16h;
typedef __attribute__((ext_vector_type(2)))  _Float16 v2h;
typedef __attribute__((ext_vector_type(8)))  float    v8f;

#define B_     8
#define S_     4096
#define HID_   640
#define KV_    77
#define AU_    16
#define CROSS_ 768
#define HEADS_ 10
#define DH_    64

static __device__ inline v8f vzero8() {
    v8f z = {0.f, 0.f, 0.f, 0.f, 0.f, 0.f, 0.f, 0.f};
    return z;
}

static __device__ inline v8f wmma_f16(v16h a, v16h b, v8f c) {
    return __builtin_amdgcn_wmma_f32_16x16x32_f16(
        false, a, false, b, (short)0, c, false, false);
}

// ---- fragment-order LDS layout helpers -------------------------------------
// Each 16x32 (A) or 32x16 (B) fragment lives in LDS as [32 lanes][16 halves];
// a fragment load is one contiguous 32B read per lane (2x ds_load_b128).

// A fragment (16x32 f16): element (m,k) -> lane = m + 16*bit3(k),
//                         slot = (k&7) | ((k&16)>>1).  k even.
static __device__ inline void a_frag_store_pair_h(_Float16* frag, int m, int k, v2h t) {
    int lane = m + ((k & 8) << 1);
    int slot = (k & 7) | ((k & 16) >> 1);
    *(v2h*)(frag + lane * 16 + slot) = t;
}
static __device__ inline void a_frag_store_pair(_Float16* frag, int m, int k,
                                                float v0, float v1) {
    v2h t = {(_Float16)v0, (_Float16)v1};
    a_frag_store_pair_h(frag, m, k, t);
}

// B fragment (32x16 f16): element (k,n) -> lane = n + 16*bit3(k),
//                         slot = (k&7) | ((k&16)>>1).  k even.
static __device__ inline void b_frag_store_pair_h(_Float16* frag, int k, int n, v2h t) {
    *(v2h*)(frag + (n + ((k & 8) << 1)) * 16 + ((k & 7) | ((k & 16) >> 1))) = t;
}
static __device__ inline void b_frag_store_pair(_Float16* frag, int k, int n,
                                                float v0, float v1) {
    v2h t = {(_Float16)v0, (_Float16)v1};
    b_frag_store_pair_h(frag, k, n, t);
}

static __device__ inline v16h frag_load(const _Float16* frag) {
    int lane = threadIdx.x & 31;
    return *(const v16h*)(frag + lane * 16);   // 32B contiguous -> 2x ds_load_b128
}

// ---------------------------------------------------------------------------
// Generic GEMM: Y[M,N] = X[M,K] * W[K,N] (+bias[N]) (+resid[M,N])
// f32 in/out, f16 WMMA compute. Block tile 128x128, 8 waves, wave tile 32x64.
// Double-buffered LDS software pipeline: one barrier per K-step; next tile is
// prefetched from global into registers (pre-converted to packed f16) while
// WMMAs consume the current LDS buffer.
// Requires K % 32 == 0, N % 128 == 0. M guarded via clamped pointer + select.
// ---------------------------------------------------------------------------
__global__ __launch_bounds__(256) void wmma_gemm_kernel(
    const float* __restrict__ X, const float* __restrict__ W,
    const float* __restrict__ bias, const float* __restrict__ resid,
    float* __restrict__ Y, int M, int K, int N)
{
    __shared__ __align__(32) _Float16 Afrag[2][8][512];   // 2 buf x 8 M-subtiles
    __shared__ __align__(32) _Float16 Bfrag[2][8][512];   // 2 buf x 8 N-subtiles

    const int tid  = threadIdx.x;
    const int lane = tid & 31;
    const int wave = tid >> 5;
    const int wms  = (wave & 3) * 2;     // A subtile base (2 per wave)
    const int wns  = (wave >> 2) * 4;    // B subtile base (4 per wave)
    const int m0 = blockIdx.x * 128;
    const int n0 = blockIdx.y * 128;

    v8f acc[2][4];
#pragma unroll
    for (int i = 0; i < 2; ++i)
#pragma unroll
        for (int j = 0; j < 4; ++j) acc[i][j] = vzero8();

    // A staging geometry: 128 rows, 2 thr/row, 16 k each
    const int arow = tid >> 1;
    const int acol = (tid & 1) * 16;
    const int asub = arow >> 4;
    const int am   = arow & 15;
    const int gm   = m0 + arow;
    const bool avalid = gm < M;
    const float* abase = X + (size_t)(avalid ? gm : 0) * K + acol;  // clamped

    // B staging geometry: unit = (k-pair, 4n); thread covers k-pairs bk0, bk0+16
    const int bk0 = (tid >> 5) * 2;      // 0,2,..,14
    const int nt  = (tid & 31) * 4;      // 0,4,..,124
    const float* bbase = W + n0 + nt;

    v2h apre[8], bpre0[4], bpre1[4];

    auto prefetch = [&](int kt) {
#pragma unroll
        for (int i = 0; i < 4; ++i) {
            float4 v = *(const float4*)(abase + kt + 4 * i);
            if (!avalid) v = make_float4(0.f, 0.f, 0.f, 0.f);
            apre[2 * i]     = v2h{(_Float16)v.x, (_Float16)v.y};
            apre[2 * i + 1] = v2h{(_Float16)v.z, (_Float16)v.w};
        }
        {
            const float* p = bbase + (size_t)(kt + bk0) * N;
            float4 r0 = *(const float4*)p;
            float4 r1 = *(const float4*)(p + N);
            bpre0[0] = v2h{(_Float16)r0.x, (_Float16)r1.x};
            bpre0[1] = v2h{(_Float16)r0.y, (_Float16)r1.y};
            bpre0[2] = v2h{(_Float16)r0.z, (_Float16)r1.z};
            bpre0[3] = v2h{(_Float16)r0.w, (_Float16)r1.w};
        }
        {
            const float* p = bbase + (size_t)(kt + bk0 + 16) * N;
            float4 r0 = *(const float4*)p;
            float4 r1 = *(const float4*)(p + N);
            bpre1[0] = v2h{(_Float16)r0.x, (_Float16)r1.x};
            bpre1[1] = v2h{(_Float16)r0.y, (_Float16)r1.y};
            bpre1[2] = v2h{(_Float16)r0.z, (_Float16)r1.z};
            bpre1[3] = v2h{(_Float16)r0.w, (_Float16)r1.w};
        }
    };
    auto stage = [&](int buf) {
#pragma unroll
        for (int i = 0; i < 4; ++i) {
            a_frag_store_pair_h(Afrag[buf][asub], am, acol + 4 * i,     apre[2 * i]);
            a_frag_store_pair_h(Afrag[buf][asub], am, acol + 4 * i + 2, apre[2 * i + 1]);
        }
        _Float16* bf = Bfrag[buf][nt >> 4];
        const int nnl = nt & 15;
#pragma unroll
        for (int j = 0; j < 4; ++j) {
            b_frag_store_pair_h(bf, bk0,      nnl + j, bpre0[j]);
            b_frag_store_pair_h(bf, bk0 + 16, nnl + j, bpre1[j]);
        }
    };

    prefetch(0);
    stage(0);
    __syncthreads();

    const int nk = K >> 5;
    for (int it = 0; it < nk; ++it) {
        const int cur = it & 1;
        if (it + 1 < nk) prefetch((it + 1) << 5);
        v16h a0 = frag_load(Afrag[cur][wms]);
        v16h a1 = frag_load(Afrag[cur][wms + 1]);
#pragma unroll
        for (int j = 0; j < 4; ++j) {
            v16h bb = frag_load(Bfrag[cur][wns + j]);
            acc[0][j] = wmma_f16(a0, bb, acc[0][j]);
            acc[1][j] = wmma_f16(a1, bb, acc[1][j]);
        }
        if (it + 1 < nk) stage(cur ^ 1);
        __syncthreads();
    }

    // ---- epilogue
    const int hi = lane >> 4;
    const int nn = lane & 15;
    const int wm = (wave & 3) * 32;
    const int wn = (wave >> 2) * 64;
#pragma unroll
    for (int i = 0; i < 2; ++i)
#pragma unroll
        for (int j = 0; j < 4; ++j)
#pragma unroll
            for (int r = 0; r < 8; ++r) {
                int row = m0 + wm + i * 16 + r + 8 * hi;
                int col = n0 + wn + j * 16 + nn;
                if (row < M) {
                    float v = acc[i][j][r];
                    if (bias)  v += bias[col];
                    if (resid) v += resid[(size_t)row * N + col];
                    Y[(size_t)row * N + col] = v;
                }
            }
}

// ---------------------------------------------------------------------------
// Fused attention: block = (64 S-rows, head, batch), 4 waves, 128 threads.
// All operands staged in fragment order; writes out + gate*au_out in place.
// LDS overlay (60KB):
//  [0,20480)      phase1-2: Qfrag(8K) KtFrag(10K) auKtFrag(2K)
//                 phase4-5: VsFrag(12K) auVsFrag(4K)
//  [20480,45056)  Sc f32 (20K) + AuSc f32 (4K)
//  [45056,61440)  Pfrag (12K) + AuFrag (4K)
// ---------------------------------------------------------------------------
__global__ __launch_bounds__(128) void attn_kernel(
    float* __restrict__ q,              // [B,S,HID]  in: Q, out: attn
    const float* __restrict__ kproj,    // [B,KV,HID]
    const float* __restrict__ vproj,    // [B,KV,HID]
    const float* __restrict__ aukproj,  // [B,AU,HID]
    const float* __restrict__ auvproj,  // [B,AU,HID]
    const float* __restrict__ temp_p,
    const float* __restrict__ gate_p)
{
    __shared__ __align__(32) char smem[61440];
    _Float16* Qfrag    = (_Float16*)(smem);            // [4 waves][2 ksub][512]
    _Float16* KtFrag   = (_Float16*)(smem + 8192);     // [2 ksub][5 nsub][512]
    _Float16* auKtFrag = (_Float16*)(smem + 18432);    // [2 ksub][512]
    _Float16* VsFrag   = (_Float16*)(smem);            // [3 ksub][4 nsub][512]
    _Float16* auVsFrag = (_Float16*)(smem + 12288);    // [4 nsub][512]
    float*    Sc       = (float*)(smem + 20480);       // [64 rows][80]
    float*    AuSc     = (float*)(smem + 40960);       // [64 rows][16]
    _Float16* Pfrag    = (_Float16*)(smem + 45056);    // [4 waves][3 ksub][512]
    _Float16* AuFrag   = (_Float16*)(smem + 57344);    // [4 waves][512]

    const int schunk = blockIdx.x * 64;
    const int h = blockIdx.y;
    const int b = blockIdx.z;
    const int tid  = threadIdx.x;
    const int lane = tid & 31;
    const int wave = tid >> 5;

    // ---- phase 1: stage Q, K^T, auK^T in fragment order
    {
        int row = tid >> 1, dbase = (tid & 1) * 32;
        int w = row >> 4, m = row & 15;
        const float* src = q + ((size_t)b * S_ + schunk + row) * HID_ + h * DH_ + dbase;
        _Float16* qf = Qfrag + (w * 2 + (dbase >> 5)) * 512;
#pragma unroll
        for (int i = 0; i < 32; i += 4) {
            float4 v = *(const float4*)(src + i);
            a_frag_store_pair(qf, m, i,     v.x, v.y);
            a_frag_store_pair(qf, m, i + 2, v.z, v.w);
        }
    }
    for (int u = tid; u < 32 * 80; u += 128) {           // K^T: d-pairs x kv
        int d2 = u & 31, kv = u >> 5;
        int d = d2 * 2;
        float2 v = make_float2(0.f, 0.f);
        if (kv < KV_)
            v = *(const float2*)(kproj + ((size_t)b * KV_ + kv) * HID_ + h * DH_ + d);
        b_frag_store_pair(KtFrag + ((d >> 5) * 5 + (kv >> 4)) * 512,
                          d & 31, kv & 15, v.x, v.y);
    }
    for (int u = tid; u < 32 * 16; u += 128) {           // auK^T
        int d2 = u & 31, kv = u >> 5;
        int d = d2 * 2;
        float2 v = *(const float2*)(aukproj + ((size_t)b * AU_ + kv) * HID_ + h * DH_ + d);
        b_frag_store_pair(auKtFrag + (d >> 5) * 512, d & 31, kv, v.x, v.y);
    }
    __syncthreads();

    // ---- phase 2: scores (16x80 per wave) + AU scores (16x16)
    const int hi = lane >> 4;
    const int nn = lane & 15;
    v16h aq0 = frag_load(Qfrag + (wave * 2 + 0) * 512);
    v16h aq1 = frag_load(Qfrag + (wave * 2 + 1) * 512);
#pragma unroll
    for (int t = 0; t < 5; ++t) {
        v8f sc = vzero8();
        sc = wmma_f16(aq0, frag_load(KtFrag + t * 512),       sc);
        sc = wmma_f16(aq1, frag_load(KtFrag + (5 + t) * 512), sc);
#pragma unroll
        for (int r = 0; r < 8; ++r)
            Sc[(wave * 16 + r + 8 * hi) * 80 + t * 16 + nn] = sc[r] * 0.125f;
    }
    {
        v8f sc = vzero8();
        sc = wmma_f16(aq0, frag_load(auKtFrag),       sc);
        sc = wmma_f16(aq1, frag_load(auKtFrag + 512), sc);
#pragma unroll
        for (int r = 0; r < 8; ++r)
            AuSc[(wave * 16 + r + 8 * hi) * 16 + nn] = sc[r] * 0.125f;
    }
    __syncthreads();

    // ---- phase 3 (tid<64): softmax + AU mask -> fragment-order f16
    // ---- phase 4 (tid>=64): stage V / auV fragments (overlaid region)
    if (tid < 64) {
        const int w = tid >> 4, m = tid & 15;
        float* row = Sc + tid * 80;
        float mx = -1e30f;
        for (int k = 0; k < KV_; ++k) mx = fmaxf(mx, row[k]);
        float sum = 0.f;
        for (int k = 0; k < KV_; ++k) {
            float e = __expf(row[k] - mx);
            row[k] = e;
            sum += e;
        }
        float inv = 1.f / sum;
        _Float16* pf = Pfrag + w * 3 * 512;
        for (int k = 0; k < 96; k += 2) {
            float p0 = (k     < KV_) ? row[k]     * inv : 0.f;
            float p1 = (k + 1 < KV_) ? row[k + 1] * inv : 0.f;
            a_frag_store_pair(pf + (k >> 5) * 512, m, k & 31, p0, p1);
        }
        // gaussian face prior + sigmoid gate
        int s  = schunk + tid;
        int py = s >> 6, px = s & 63;
        float ly = -1.f + 2.f * (float)py / 63.f;
        float lx = -1.f + 2.f * (float)px / 63.f;
        float g  = __expf(-(lx * lx + ly * ly) * (1.f / (2.f * 0.55f * 0.55f)));
        float invT = 1.f / (fabsf(temp_p[0]) + 1e-6f);
        float* arow = AuSc + tid * 16;
        _Float16* af = AuFrag + w * 512;
        for (int k = 0; k < 32; k += 2) {
            float v0 = 0.f, v1 = 0.f;
            if (k < AU_)
                v0 = (1.f / (1.f + __expf(-arow[k] * invT))) * g * 0.9f + 0.1f;
            if (k + 1 < AU_)
                v1 = (1.f / (1.f + __expf(-arow[k + 1] * invT))) * g * 0.9f + 0.1f;
            a_frag_store_pair(af, m, k, v0, v1);
        }
    } else {
        const int t2 = tid - 64;
        for (int u = t2; u < 48 * 64; u += 64) {         // V: kv-pairs x d
            int d = u & 63, kv = (u >> 6) * 2;
            size_t base = ((size_t)b * KV_ + kv) * HID_ + h * DH_ + d;
            float v0 = (kv     < KV_) ? vproj[base]        : 0.f;
            float v1 = (kv + 1 < KV_) ? vproj[base + HID_] : 0.f;
            b_frag_store_pair(VsFrag + ((kv >> 5) * 4 + (d >> 4)) * 512,
                              kv & 31, d & 15, v0, v1);
        }
        for (int u = t2; u < 16 * 64; u += 64) {         // auV
            int d = u & 63, kv = (u >> 6) * 2;
            size_t base = ((size_t)b * AU_ + kv) * HID_ + h * DH_ + d;
            float v0 = (kv     < AU_) ? auvproj[base]        : 0.f;
            float v1 = (kv + 1 < AU_) ? auvproj[base + HID_] : 0.f;
            b_frag_store_pair(auVsFrag + (d >> 4) * 512, kv, d & 15, v0, v1);
        }
    }
    __syncthreads();

    // ---- phase 5: out = P @ V + gate * (mask @ auV)
    const float gate = gate_p[0];
    v16h p0 = frag_load(Pfrag + (wave * 3 + 0) * 512);
    v16h p1 = frag_load(Pfrag + (wave * 3 + 1) * 512);
    v16h p2 = frag_load(Pfrag + (wave * 3 + 2) * 512);
    v16h am = frag_load(AuFrag + wave * 512);
#pragma unroll
    for (int t = 0; t < 4; ++t) {
        v8f o = vzero8();
        o = wmma_f16(p0, frag_load(VsFrag + (0 * 4 + t) * 512), o);
        o = wmma_f16(p1, frag_load(VsFrag + (1 * 4 + t) * 512), o);
        o = wmma_f16(p2, frag_load(VsFrag + (2 * 4 + t) * 512), o);
        v8f ao = vzero8();
        ao = wmma_f16(am, frag_load(auVsFrag + t * 512), ao);
#pragma unroll
        for (int r = 0; r < 8; ++r) {
            size_t off = ((size_t)b * S_ + schunk + wave * 16 + r + 8 * hi) * HID_
                         + h * DH_ + t * 16 + nn;
            q[off] = o[r] + gate * ao[r];
        }
    }
}

extern "C" void kernel_launch(void* const* d_in, const int* in_sizes, int n_in,
                              void* d_out, int out_size, void* d_ws, size_t ws_size,
                              hipStream_t stream) {
    const float* hidden = (const float*)d_in[0];
    const float* enc    = (const float*)d_in[1];
    const float* au     = (const float*)d_in[2];
    const float* Wq     = (const float*)d_in[3];
    const float* Wk     = (const float*)d_in[4];
    const float* Wv     = (const float*)d_in[5];
    const float* Wauk   = (const float*)d_in[6];
    const float* Wauv   = (const float*)d_in[7];
    const float* Wout   = (const float*)d_in[8];
    const float* bout   = (const float*)d_in[9];
    const float* temp   = (const float*)d_in[10];
    const float* gate   = (const float*)d_in[11];
    float* out = (float*)d_out;

    float* ws     = (float*)d_ws;
    float* qbuf   = ws;                                  // 32768*640
    float* kbuf   = qbuf   + (size_t)B_ * S_ * HID_;     // 616*640
    float* vbuf   = kbuf   + (size_t)B_ * KV_ * HID_;    // 616*640
    float* aukbuf = vbuf   + (size_t)B_ * KV_ * HID_;    // 128*640
    float* auvbuf = aukbuf + (size_t)B_ * AU_ * HID_;    // 128*640

    const dim3 blk(256);
    wmma_gemm_kernel<<<dim3(256, 5), blk, 0, stream>>>(
        hidden, Wq, nullptr, nullptr, qbuf, B_ * S_, HID_, HID_);
    wmma_gemm_kernel<<<dim3(5, 5), blk, 0, stream>>>(
        enc, Wk, nullptr, nullptr, kbuf, B_ * KV_, CROSS_, HID_);
    wmma_gemm_kernel<<<dim3(5, 5), blk, 0, stream>>>(
        enc, Wv, nullptr, nullptr, vbuf, B_ * KV_, CROSS_, HID_);
    wmma_gemm_kernel<<<dim3(1, 5), blk, 0, stream>>>(
        au, Wauk, nullptr, nullptr, aukbuf, B_ * AU_, CROSS_, HID_);
    wmma_gemm_kernel<<<dim3(1, 5), blk, 0, stream>>>(
        au, Wauv, nullptr, nullptr, auvbuf, B_ * AU_, CROSS_, HID_);
    attn_kernel<<<dim3(S_ / 64, HEADS_, B_), dim3(128), 0, stream>>>(
        qbuf, kbuf, vbuf, aukbuf, auvbuf, temp, gate);
    wmma_gemm_kernel<<<dim3(256, 5), blk, 0, stream>>>(
        qbuf, Wout, bout, hidden, out, B_ * S_, HID_, HID_);
}
